// BitLinear_16054587753064
// MI455X (gfx1250) — 8-bit matrix operand order repaired and checked by reading, compiled, not run on hardware
//
#include <hip/hip_runtime.h>
#include <hip/hip_bf16.h>
#include <stdint.h>

#define TOKENS 8192
#define IN_F   4096
#define OUT_F  4096

#define BM 128
#define BN 128
#define BK 64
#define LDS_STRIDE 80

typedef __attribute__((ext_vector_type(8))) int   v8i;
typedef __attribute__((ext_vector_type(4))) int   v4i;
typedef __attribute__((ext_vector_type(2))) int   v2i;
typedef __attribute__((ext_vector_type(4))) float v4f;


__device__ __forceinline__ void async_ld16(uint32_t lds_off, uint32_t voff, const void* sbase) {
    asm volatile("global_load_async_to_lds_b128 %0, %1, %2"
                 :: "v"(lds_off), "v"(voff), "s"(sbase)
                 : "memory");
}

__device__ __forceinline__ void wait_async0() {
    asm volatile("s_wait_asynccnt 0x0" ::: "memory");
}


__device__ __forceinline__ int q8pack(v4f f, float inv, float lo, float hi) {
    int a = (int)fminf(hi, fmaxf(lo, rintf(f.x * inv)));
    int b = (int)fminf(hi, fmaxf(lo, rintf(f.y * inv)));
    int c = (int)fminf(hi, fmaxf(lo, rintf(f.z * inv)));
    int d = (int)fminf(hi, fmaxf(lo, rintf(f.w * inv)));
    return (a & 0xFF) | ((b & 0xFF) << 8) | ((c & 0xFF) << 16) | ((d & 0xFF) << 24);
}

__global__ void quantize_kernel(const float* __restrict__ src,
                                const float* __restrict__ scale_p,
                                v4i* __restrict__ dstq,
                                float lo, float hi) {
    const float inv = 1.0f / scale_p[0];
    const int i = blockIdx.x * blockDim.x + threadIdx.x;
    const v4f* s = (const v4f*)src + (size_t)i * 4;
    v4f f0 = __builtin_nontemporal_load(s + 0);
    v4f f1 = __builtin_nontemporal_load(s + 1);
    v4f f2 = __builtin_nontemporal_load(s + 2);
    v4f f3 = __builtin_nontemporal_load(s + 3);
    v4i o;
    o.x = q8pack(f0, inv, lo, hi);
    o.y = q8pack(f1, inv, lo, hi);
    o.z = q8pack(f2, inv, lo, hi);
    o.w = q8pack(f3, inv, lo, hi);
    dstq[i] = o;
}


__device__ __forceinline__ void compute_tile(const signed char* __restrict__ sAbuf,
                                             const signed char* __restrict__ sBbuf,
                                             int arow, int akoff, int bcol, int bkoff,
                                             v8i acc[2][4]) {
    v8i af[2];
    #pragma unroll
    for (int mt = 0; mt < 2; ++mt) {
        const signed char* ab = sAbuf + (arow + mt * 16) * LDS_STRIDE + akoff;
        v2i q0 = *(const v2i*)(ab);
        v2i q1 = *(const v2i*)(ab + 8);
        v2i q2 = *(const v2i*)(ab + 32);
        v2i q3 = *(const v2i*)(ab + 40);
        af[mt] = (v8i){q0.x, q0.y, q1.x, q1.y, q2.x, q2.y, q3.x, q3.y};
    }
    v8i bf[4];
    #pragma unroll
    for (int nt = 0; nt < 4; ++nt) {
        const signed char* bb = sBbuf + (bcol + nt * 16) * LDS_STRIDE + bkoff;
        v4i p0 = *(const v4i*)(bb);
        v4i p1 = *(const v4i*)(bb + 32);
        bf[nt] = (v8i){p0.x, p0.y, p0.z, p0.w, p1.x, p1.y, p1.z, p1.w};
    }
    #pragma unroll
    for (int mt = 0; mt < 2; ++mt)
        #pragma unroll
        for (int nt = 0; nt < 4; ++nt)
            acc[mt][nt] = __builtin_amdgcn_wmma_i32_16x16x64_iu8(
                 true, af[mt],
                 true, bf[nt],
                acc[mt][nt],
                 false,  false);
}

__global__ __launch_bounds__(256)
void bitlinear_wmma_kernel(const signed char* __restrict__ Aq,
                           const signed char* __restrict__ Bq,
                           const float* __restrict__ wscale_p,
                           const float* __restrict__ xscale_p,
                           float* __restrict__ out)
{
    __shared__ signed char sA[2][BM * LDS_STRIDE];
    __shared__ signed char sB[2][BN * LDS_STRIDE];

    const int tid  = threadIdx.x;
    const int wave = tid >> 5;
    const int lane = tid & 31;
    const int bm = blockIdx.y * BM;
    const int bn = blockIdx.x * BN;

    const int wm = (wave >> 1) * 32;
    const int wn = (wave & 1) * 64;

    const int c0 = tid, c1 = tid + 256;
    const int ar0 = c0 >> 2, as0 = (c0 & 3) * 16;
    const int ar1 = c1 >> 2, as1 = (c1 & 3) * 16;

    const uint32_t vA0 = (uint32_t)(bm + ar0) * IN_F + as0;
    const uint32_t vA1 = (uint32_t)(bm + ar1) * IN_F + as1;
    const uint32_t vB0 = (uint32_t)(bn + ar0) * IN_F + as0;
    const uint32_t vB1 = (uint32_t)(bn + ar1) * IN_F + as1;

    uint32_t lA[2], lA2[2], lB[2], lB2[2];
    #pragma unroll
    for (int p = 0; p < 2; ++p) {
        lA[p]  = (uint32_t)(uintptr_t)(&sA[p][ar0 * LDS_STRIDE + as0]);
        lA2[p] = (uint32_t)(uintptr_t)(&sA[p][ar1 * LDS_STRIDE + as1]);
        lB[p]  = (uint32_t)(uintptr_t)(&sB[p][ar0 * LDS_STRIDE + as0]);
        lB2[p] = (uint32_t)(uintptr_t)(&sB[p][ar1 * LDS_STRIDE + as1]);
    }

    v8i acc[2][4];
    #pragma unroll
    for (int mt = 0; mt < 2; ++mt)
        #pragma unroll
        for (int nt = 0; nt < 4; ++nt)
            acc[mt][nt] = (v8i)0;

    const int arow  = wm + (lane & 15);
    const int akoff = (lane >> 4) * 16;
    const int bcol  = wn + (lane & 15);
    const int bkoff = (lane >> 4) * 16;

    async_ld16(lA[0],  vA0, Aq);
    async_ld16(lA2[0], vA1, Aq);
    async_ld16(lB[0],  vB0, Bq);
    async_ld16(lB2[0], vB1, Bq);
    wait_async0();
    __syncthreads();

    for (int kk = 0; kk < IN_F; kk += 2 * BK) {
        {
            const signed char* ka = Aq + kk + BK;
            const signed char* kb = Bq + kk + BK;
            async_ld16(lA[1],  vA0, ka);
            async_ld16(lA2[1], vA1, ka);
            async_ld16(lB[1],  vB0, kb);
            async_ld16(lB2[1], vB1, kb);
        }

        compute_tile(sA[0], sB[0], arow, akoff, bcol, bkoff, acc);

        wait_async0();
        __syncthreads();

        if (kk + 2 * BK < IN_F) {
            const signed char* ka = Aq + kk + 2 * BK;
            const signed char* kb = Bq + kk + 2 * BK;
            async_ld16(lA[0],  vA0, ka);
            async_ld16(lA2[0], vA1, ka);
            async_ld16(lB[0],  vB0, kb);
            async_ld16(lB2[0], vB1, kb);
        }

        compute_tile(sA[1], sB[1], arow, akoff, bcol, bkoff, acc);

        wait_async0();
        __syncthreads();
    }

    const float dq = wscale_p[0] * xscale_p[0];

    #pragma unroll
    for (int mt = 0; mt < 2; ++mt) {
        #pragma unroll
        for (int nt = 0; nt < 4; ++nt) {
            const int gm = bm + wm + mt * 16 + (lane >> 4) * 8;
            const int gn = bn + wn + nt * 16 + (lane & 15);
            float* po = out + (size_t)gm * OUT_F + gn;
            #pragma unroll
            for (int r = 0; r < 8; ++r)
                __builtin_nontemporal_store((float)acc[mt][nt][r] * dq,
                                            po + (size_t)r * OUT_F);
        }
    }
}


extern "C" void kernel_launch(void* const* d_in, const int* in_sizes, int n_in,
                              void* d_out, int out_size, void* d_ws, size_t ws_size,
                              hipStream_t stream) {
    (void)in_sizes; (void)n_in; (void)out_size; (void)ws_size;
    const float* x  = (const float*)d_in[0];
    const float* w  = (const float*)d_in[1];
    const float* ws = (const float*)d_in[2];
    const float* xs = (const float*)d_in[3];
    float* out = (float*)d_out;

    signed char* xq = (signed char*)d_ws;
    signed char* wq = xq + (size_t)TOKENS * IN_F;

    quantize_kernel<<<(TOKENS * IN_F / 16) / 256, 256, 0, stream>>>(
        x, xs, (v4i*)xq, -128.0f, 127.0f);
    quantize_kernel<<<(OUT_F * IN_F / 16) / 256, 256, 0, stream>>>(
        w, ws, (v4i*)wq, -1.0f, 1.0f);

    dim3 grid(OUT_F / BN, TOKENS / BM);
    bitlinear_wmma_kernel<<<grid, 256, 0, stream>>>(xq, wq, ws, xs, out);
}
